// EquivariantDecoder_50491635531841
// MI455X (gfx1250) — matrix operand order repaired and checked by reading, not compiled, not run on hardware
//
#include <hip/hip_runtime.h>

#define NA 512          // atoms (sequence length)
#define LATD 128
#define HD 256          // hidden
#define NLAY 3
#define NHEAD 4
#define DHEAD 64
#define EMBD 64
#define ZIND (LATD + EMBD)   // 192
#define FFD (4 * HD)         // 1024

typedef __bf16 v16bf __attribute__((ext_vector_type(16)));
typedef __bf16 v8bf  __attribute__((ext_vector_type(8)));
typedef float  v8f   __attribute__((ext_vector_type(8)));

__device__ __forceinline__ float siluf(float x) { return x / (1.0f + __expf(-x)); }
__device__ __forceinline__ float softplusf(float x) { return (x > 20.0f) ? x : log1pf(__expf(x)); }

// ---- WMMA fragment loaders (bf16, 16x16x32), contiguous k order, same for A and B ----
// A (16x32, MxK) row-major: lane r=lane%16 row, half=lane/16: K = half*16 + e (contiguous)
__device__ __forceinline__ v16bf load_frag_a(const __bf16* base, int stride, int row0,
                                             int k0, int r, int half) {
  const __bf16* p = base + (size_t)(row0 + r) * stride + k0 + (half << 4);
  v8bf lo = *(const v8bf*)(p);
  v8bf hi = *(const v8bf*)(p + 8);
  v16bf a;
#pragma unroll
  for (int i = 0; i < 8; ++i) { a[i] = lo[i]; a[i + 8] = hi[i]; }
  return a;
}

// B (32x16, KxN) from W stored row-major (N x K): column n = col0 + r, K = half*16 + e (contiguous)
__device__ __forceinline__ v16bf load_frag_bt(const __bf16* base, int stride, int col0,
                                              int k0, int r, int half) {
  const __bf16* p = base + (size_t)(col0 + r) * stride + k0 + (half << 4);
  v8bf lo = *(const v8bf*)(p);
  v8bf hi = *(const v8bf*)(p + 8);
  v16bf b;
#pragma unroll
  for (int i = 0; i < 8; ++i) { b[i] = lo[i]; b[i + 8] = hi[i]; }
  return b;
}

// ---------------------------------------------------------------------------
// Generic WMMA GEMM: C = act(alpha * A @ op(B) + bias + bias2 + resid)
// A: (M x K) bf16 row-major (lda). BT=true: B is W (N x K) row-major -> A@W^T.
// BT=false: B is (K x N) row-major. One wave per 16x16 tile, 4 waves/block in N.
// blockIdx.z batches (heads) with element strides sA/sB/sC.
// ACT: 0 none, 1 silu, 2 relu.
// ---------------------------------------------------------------------------
template <int ACT, bool BT>
__global__ __launch_bounds__(128) void gemm_kernel(
    const __bf16* __restrict__ A, int lda, long sA,
    const __bf16* __restrict__ B, int ldb, long sB,
    const float* __restrict__ bias, const float* __restrict__ bias2,
    const float* __restrict__ resid, int ldr,
    float* __restrict__ Cf, int ldc,
    __bf16* __restrict__ Cb, int ldcb, long sC,
    int N, int K, float alpha) {
  const int z = blockIdx.z;
  A += (size_t)z * sA;
  B += (size_t)z * sB;
  const int lane = threadIdx.x & 31;
  const int r = lane & 15;
  const int half = lane >> 4;
  const int m0 = blockIdx.y << 4;
  const int n0 = (blockIdx.x << 6) + ((threadIdx.x >> 5) << 4);
  if (n0 >= N) return;  // wave-uniform: EXEC stays all-ones for WMMA

  v8f c = {0.f, 0.f, 0.f, 0.f, 0.f, 0.f, 0.f, 0.f};
  for (int k0 = 0; k0 < K; k0 += 32) {
    v16bf a = load_frag_a(A, lda, m0, k0, r, half);
    v16bf b;
    if (BT) {
      b = load_frag_bt(B, ldb, n0, k0, r, half);
    } else {
      const __bf16* p = B + (size_t)(k0 + (half << 4)) * ldb + n0 + r;
#pragma unroll
      for (int e = 0; e < 16; ++e) b[e] = p[(size_t)e * ldb];
    }
    c = __builtin_amdgcn_wmma_f32_16x16x32_bf16(false, a, false, b, (short)0, c,
                                                false, false);
  }
  const int n = n0 + r;
#pragma unroll
  for (int i = 0; i < 8; ++i) {
    const int m = m0 + i + (half << 3);
    float v = c[i] * alpha;
    if (bias)  v += bias[n];
    if (bias2) v += bias2[n];
    if (resid) v += resid[(size_t)m * ldr + n];
    if (ACT == 1) v = siluf(v);
    if (ACT == 2) v = fmaxf(v, 0.0f);
    if (Cf) Cf[(size_t)z * sC + (size_t)m * ldc + n] = v;
    if (Cb) Cb[(size_t)z * sC + (size_t)m * ldcb + n] = (__bf16)v;
  }
}

template <int ACT, bool BT>
static void launch_gemm(hipStream_t st,
                        const __bf16* A, int lda, long sA,
                        const __bf16* B, int ldb, long sB,
                        const float* bias, const float* bias2,
                        const float* resid, int ldr,
                        float* Cf, int ldc, __bf16* Cb, int ldcb, long sC,
                        int M, int N, int K, int Z, float alpha = 1.0f) {
  dim3 grid(N / 64, M / 16, Z);
  gemm_kernel<ACT, BT><<<grid, 128, 0, st>>>(A, lda, sA, B, ldb, sB, bias, bias2,
                                             resid, ldr, Cf, ldc, Cb, ldcb, sC,
                                             N, K, alpha);
}

// ---------------------------------------------------------------------------
// Fused dist-head pair kernel. Block = one 16x16 (i,j) tile = 256 pair rows.
// Stage A = silu(ai_i + aj_j (+b1 folded into ai)) in LDS as bf16, then a
// 256x256x256 WMMA GEMM vs dp_w2^T, epilogue silu(.+b2)*w3 reduced per pair
// row via deterministic shfl_xor, finally softplus(sum + b3).
// ---------------------------------------------------------------------------
__global__ __launch_bounds__(256) void pair_kernel(
    const float* __restrict__ ai, const float* __restrict__ aj,
    const __bf16* __restrict__ w2, const float* __restrict__ b2,
    const float* __restrict__ w3, const float* __restrict__ b3,
    float* __restrict__ dist_raw) {
  extern __shared__ __bf16 As[];         // 256 rows x (HD+8) bf16
  __shared__ float part[256][16];        // per-pair-row partial dots, per n-tile
  const int APAD = HD + 8;               // 264: pads LDS banks for frag loads
  const int t = threadIdx.x;
  const int i0 = blockIdx.y << 4, j0 = blockIdx.x << 4;
  {
    const int ii = t >> 4, jj = t & 15;
    const float* ar = ai + (size_t)(i0 + ii) * HD;
    const float* br = aj + (size_t)(j0 + jj) * HD;
#pragma unroll 4
    for (int k = 0; k < HD; ++k)
      As[t * APAD + k] = (__bf16)siluf(ar[k] + br[k]);
  }
  __syncthreads();

  const int wave = t >> 5, lane = t & 31, r = lane & 15, half = lane >> 4;
  for (int tt = 0; tt < 32; ++tt) {
    const int tile = tt * 8 + wave;         // 256 tiles over 8 waves
    const int m0 = (tile >> 4) << 4;        // pair-row tile
    const int n0 = (tile & 15) << 4;        // feature tile
    v8f c = {0.f, 0.f, 0.f, 0.f, 0.f, 0.f, 0.f, 0.f};
    for (int k0 = 0; k0 < HD; k0 += 32) {
      const __bf16* ap = As + (size_t)(m0 + r) * APAD + k0 + (half << 4);
      v8bf alo = *(const v8bf*)(ap);
      v8bf ahi = *(const v8bf*)(ap + 8);
      v16bf a;
#pragma unroll
      for (int i = 0; i < 8; ++i) { a[i] = alo[i]; a[i + 8] = ahi[i]; }
      v16bf b = load_frag_bt(w2, HD, n0, k0, r, half);
      c = __builtin_amdgcn_wmma_f32_16x16x32_bf16(false, a, false, b, (short)0, c,
                                                  false, false);
    }
    const int n = n0 + r;
    const float w3n = w3[n], b2n = b2[n];
#pragma unroll
    for (int i = 0; i < 8; ++i) {
      float v = siluf(c[i] + b2n) * w3n;
      v += __shfl_xor(v, 1, 32);
      v += __shfl_xor(v, 2, 32);
      v += __shfl_xor(v, 4, 32);
      v += __shfl_xor(v, 8, 32);           // sum over the 16-lane N group
      if (r == 0) part[m0 + i + (half << 3)][tile & 15] = v;
    }
  }
  __syncthreads();
  float d = b3[0];
#pragma unroll
  for (int q = 0; q < 16; ++q) d += part[t][q];
  dist_raw[(size_t)(i0 + (t >> 4)) * NA + j0 + (t & 15)] = softplusf(d);
}

// ----------------------------- small kernels -------------------------------
__global__ void f2bf_kernel(const float* __restrict__ in, __bf16* __restrict__ out, int n) {
  int i = blockIdx.x * blockDim.x + threadIdx.x;
  if (i < n) out[i] = (__bf16)in[i];
}

__global__ void zin_kernel(const float* __restrict__ z, const int* __restrict__ at,
                           const float* __restrict__ emb, __bf16* __restrict__ zin) {
  int idx = blockIdx.x * blockDim.x + threadIdx.x;
  if (idx >= NA * ZIND) return;
  int i = idx / ZIND, c = idx % ZIND;
  float v = (c < LATD) ? z[i * LATD + c] : emb[at[i] * EMBD + (c - LATD)];
  zin[idx] = (__bf16)v;
}

// global path: g = Lin(silu(Lin(z_global, w1,b1)), w2,b2), one block of HD threads
__global__ __launch_bounds__(HD) void gp_kernel(
    const float* __restrict__ zg, const float* __restrict__ w1, const float* __restrict__ b1,
    const float* __restrict__ w2, const float* __restrict__ b2, float* __restrict__ g) {
  __shared__ float g1[HD];
  int t = threadIdx.x;
  float acc = b1[t];
  for (int k = 0; k < LATD; ++k) acc += zg[k] * w1[t * LATD + k];
  g1[t] = siluf(acc);
  __syncthreads();
  float a2 = b2[t];
  for (int k = 0; k < HD; ++k) a2 += g1[k] * w2[t * HD + k];
  g[t] = a2;
}

__global__ __launch_bounds__(HD) void ln_kernel(
    const float* __restrict__ in, const float* __restrict__ sc, const float* __restrict__ bi,
    float* __restrict__ h, __bf16* __restrict__ hb) {
  __shared__ float red[HD];
  int row = blockIdx.x, t = threadIdx.x;
  float v = in[(size_t)row * HD + t];
  red[t] = v; __syncthreads();
  for (int o = HD / 2; o > 0; o >>= 1) { if (t < o) red[t] += red[t + o]; __syncthreads(); }
  float mean = red[0] * (1.0f / HD); __syncthreads();
  float d = v - mean;
  red[t] = d * d; __syncthreads();
  for (int o = HD / 2; o > 0; o >>= 1) { if (t < o) red[t] += red[t + o]; __syncthreads(); }
  float var = red[0] * (1.0f / HD);
  float ov = d * rsqrtf(var + 1e-5f) * sc[t] + bi[t];
  h[(size_t)row * HD + t] = ov;
  hb[(size_t)row * HD + t] = (__bf16)ov;
}

__global__ __launch_bounds__(256) void softmax_kernel(
    const float* __restrict__ s, __bf16* __restrict__ p) {
  __shared__ float red[256];
  int row = blockIdx.x, t = threadIdx.x;       // rows = NHEAD*NA, cols = NA (512)
  const float* sr = s + (size_t)row * NA;
  float a = sr[t], b = sr[t + 256];
  red[t] = fmaxf(a, b); __syncthreads();
  for (int o = 128; o > 0; o >>= 1) { if (t < o) red[t] = fmaxf(red[t], red[t + o]); __syncthreads(); }
  float m = red[0]; __syncthreads();
  float ea = __expf(a - m), eb = __expf(b - m);
  red[t] = ea + eb; __syncthreads();
  for (int o = 128; o > 0; o >>= 1) { if (t < o) red[t] += red[t + o]; __syncthreads(); }
  float inv = 1.0f / red[0];
  __bf16* pr = p + (size_t)row * NA;
  pr[t] = (__bf16)(ea * inv);
  pr[t + 256] = (__bf16)(eb * inv);
}

__global__ void sym_kernel(const float* __restrict__ raw, float* __restrict__ out) {
  int idx = blockIdx.x * blockDim.x + threadIdx.x;
  if (idx >= NA * NA) return;
  int i = idx / NA, j = idx % NA;
  out[idx] = 0.5f * (raw[(size_t)i * NA + j] + raw[(size_t)j * NA + i]);
}

__global__ void coord_kernel(const float* __restrict__ hc, const float* __restrict__ w,
                             const float* __restrict__ b, float* __restrict__ out) {
  int idx = blockIdx.x * blockDim.x + threadIdx.x;
  if (idx >= NA * 3) return;
  int i = idx / 3, o = idx % 3;
  float acc = b[o];
  for (int k = 0; k < HD; ++k) acc += hc[(size_t)i * HD + k] * w[o * HD + k];
  out[idx] = acc;
}

// ---------------------------------------------------------------------------
extern "C" void kernel_launch(void* const* d_in, const int* in_sizes, int n_in,
                              void* d_out, int out_size, void* d_ws, size_t ws_size,
                              hipStream_t stream) {
  (void)in_sizes; (void)n_in; (void)out_size; (void)ws_size;
  const float* z     = (const float*)d_in[0];
  const float* zg    = (const float*)d_in[1];
  const int*   atoms = (const int*)  d_in[2];
  const float* emb   = (const float*)d_in[3];
  const float* ip_w1 = (const float*)d_in[4];
  const float* ip_b1 = (const float*)d_in[5];
  const float* ip_w2 = (const float*)d_in[6];
  const float* ip_b2 = (const float*)d_in[7];
  const float* gp_w1 = (const float*)d_in[8];
  const float* gp_b1 = (const float*)d_in[9];
  const float* gp_w2 = (const float*)d_in[10];
  const float* gp_b2 = (const float*)d_in[11];
  const float* wqkv  = (const float*)d_in[12];
  const float* bqkv  = (const float*)d_in[13];
  const float* wo    = (const float*)d_in[14];
  const float* bo    = (const float*)d_in[15];
  const float* ln1_s = (const float*)d_in[16];
  const float* ln1_b = (const float*)d_in[17];
  const float* w1    = (const float*)d_in[18];
  const float* b1    = (const float*)d_in[19];
  const float* w2    = (const float*)d_in[20];
  const float* b2    = (const float*)d_in[21];
  const float* ln2_s = (const float*)d_in[22];
  const float* ln2_b = (const float*)d_in[23];
  const float* dp_w1 = (const float*)d_in[24];
  const float* dp_b1 = (const float*)d_in[25];
  const float* dp_w2 = (const float*)d_in[26];
  const float* dp_b2 = (const float*)d_in[27];
  const float* dp_w3 = (const float*)d_in[28];
  const float* dp_b3 = (const float*)d_in[29];
  const float* cp_w1 = (const float*)d_in[30];
  const float* cp_b1 = (const float*)d_in[31];
  const float* cp_w2 = (const float*)d_in[32];
  const float* cp_b2 = (const float*)d_in[33];
  float* out_x    = (float*)d_out;            // (512,3)
  float* out_dist = (float*)d_out + NA * 3;   // (512,512)

  // --- workspace carving (256B aligned) ---
  size_t off = 0;
  auto af = [&](size_t n) -> float* {
    float* p = (float*)((char*)d_ws + off);
    off = (off + n * sizeof(float) + 255) & ~(size_t)255; return p;
  };
  auto ab = [&](size_t n) -> __bf16* {
    __bf16* p = (__bf16*)((char*)d_ws + off);
    off = (off + n * sizeof(__bf16) + 255) & ~(size_t)255; return p;
  };
  __bf16* ip_w1_b = ab(HD * ZIND);
  __bf16* ip_w2_b = ab(HD * HD);
  __bf16* wqkv_b  = ab((size_t)NLAY * 3 * HD * HD);
  __bf16* wo_b    = ab((size_t)NLAY * HD * HD);
  __bf16* w1_b    = ab((size_t)NLAY * FFD * HD);
  __bf16* w2_b    = ab((size_t)NLAY * HD * FFD);
  __bf16* dpw1_b  = ab(HD * 2 * HD);
  __bf16* dpw2_b  = ab(HD * HD);
  __bf16* cpw1_b  = ab(HD * HD);
  __bf16* zin_b   = ab((size_t)NA * ZIND);
  __bf16* t1_b    = ab((size_t)NA * HD);
  __bf16* h_b     = ab((size_t)NA * HD);
  __bf16* qkv_b   = ab((size_t)NA * 3 * HD);
  __bf16* probs_b = ab((size_t)NHEAD * NA * NA);
  __bf16* o_b     = ab((size_t)NA * HD);
  __bf16* ffn1_b  = ab((size_t)NA * FFD);
  float* g      = af(HD);
  float* h      = af((size_t)NA * HD);
  float* scores = af((size_t)NHEAD * NA * NA);
  float* preln  = af((size_t)NA * HD);
  float* ai_    = af((size_t)NA * HD);
  float* aj_    = af((size_t)NA * HD);
  float* hc     = af((size_t)NA * HD);
  float* draw   = af((size_t)NA * NA);

  // --- weight conversion to bf16 ---
  auto conv = [&](const float* s, __bf16* d, int n) {
    f2bf_kernel<<<(n + 255) / 256, 256, 0, stream>>>(s, d, n);
  };
  conv(ip_w1, ip_w1_b, HD * ZIND);
  conv(ip_w2, ip_w2_b, HD * HD);
  conv(wqkv,  wqkv_b,  NLAY * 3 * HD * HD);
  conv(wo,    wo_b,    NLAY * HD * HD);
  conv(w1,    w1_b,    NLAY * FFD * HD);
  conv(w2,    w2_b,    NLAY * HD * FFD);
  conv(dp_w1, dpw1_b,  HD * 2 * HD);
  conv(dp_w2, dpw2_b,  HD * HD);
  conv(cp_w1, cpw1_b,  HD * HD);

  zin_kernel<<<(NA * ZIND + 255) / 256, 256, 0, stream>>>(z, atoms, emb, zin_b);
  gp_kernel<<<1, HD, 0, stream>>>(zg, gp_w1, gp_b1, gp_w2, gp_b2, g);

  // input projection: h = Lin(silu(Lin(zin))) + g  (g acts as a second bias)
  launch_gemm<1, true>(stream, zin_b, ZIND, 0, ip_w1_b, ZIND, 0, ip_b1, nullptr,
                       nullptr, 0, nullptr, 0, t1_b, HD, 0, NA, HD, ZIND, 1);
  launch_gemm<0, true>(stream, t1_b, HD, 0, ip_w2_b, HD, 0, ip_b2, g,
                       nullptr, 0, h, HD, h_b, HD, 0, NA, HD, HD, 1);

  for (int l = 0; l < NLAY; ++l) {
    // QKV
    launch_gemm<0, true>(stream, h_b, HD, 0, wqkv_b + (size_t)l * 3 * HD * HD, HD, 0,
                         bqkv + l * 3 * HD, nullptr, nullptr, 0,
                         nullptr, 0, qkv_b, 3 * HD, 0, NA, 3 * HD, HD, 1);
    // scores = (Q K^T) / 8, batched over heads via z
    launch_gemm<0, true>(stream, qkv_b, 3 * HD, DHEAD, qkv_b + HD, 3 * HD, DHEAD,
                         nullptr, nullptr, nullptr, 0,
                         scores, NA, nullptr, 0, (long)NA * NA, NA, NA, DHEAD,
                         NHEAD, 0.125f);
    softmax_kernel<<<NHEAD * NA, 256, 0, stream>>>(scores, probs_b);
    // O = P @ V (B not transposed), per head
    launch_gemm<0, false>(stream, probs_b, NA, (long)NA * NA, qkv_b + 2 * HD, 3 * HD,
                          DHEAD, nullptr, nullptr, nullptr, 0,
                          nullptr, 0, o_b, HD, DHEAD, NA, DHEAD, NA, NHEAD);
    // out proj + residual, then LN1
    launch_gemm<0, true>(stream, o_b, HD, 0, wo_b + (size_t)l * HD * HD, HD, 0,
                         bo + l * HD, nullptr, h, HD,
                         preln, HD, nullptr, 0, 0, NA, HD, HD, 1);
    ln_kernel<<<NA, HD, 0, stream>>>(preln, ln1_s + l * HD, ln1_b + l * HD, h, h_b);
    // FFN
    launch_gemm<2, true>(stream, h_b, HD, 0, w1_b + (size_t)l * FFD * HD, HD, 0,
                         b1 + l * FFD, nullptr, nullptr, 0,
                         nullptr, 0, ffn1_b, FFD, 0, NA, FFD, HD, 1);
    launch_gemm<0, true>(stream, ffn1_b, FFD, 0, w2_b + (size_t)l * HD * FFD, FFD, 0,
                         b2 + l * HD, nullptr, h, HD,
                         preln, HD, nullptr, 0, 0, NA, HD, FFD, 1);
    ln_kernel<<<NA, HD, 0, stream>>>(preln, ln2_s + l * HD, ln2_b + l * HD, h, h_b);
  }

  // dist head: ai = h @ dp_w1[:, :H]^T + b1 (bias folded), aj = h @ dp_w1[:, H:]^T
  launch_gemm<0, true>(stream, h_b, HD, 0, dpw1_b, 2 * HD, 0, dp_b1, nullptr,
                       nullptr, 0, ai_, HD, nullptr, 0, 0, NA, HD, HD, 1);
  launch_gemm<0, true>(stream, h_b, HD, 0, dpw1_b + HD, 2 * HD, 0, nullptr, nullptr,
                       nullptr, 0, aj_, HD, nullptr, 0, 0, NA, HD, HD, 1);
  pair_kernel<<<dim3(NA / 16, NA / 16), 256, (HD + 8) * 256 * sizeof(__bf16), stream>>>(
      ai_, aj_, dpw2_b, dp_b2, dp_w3, dp_b3, draw);
  sym_kernel<<<(NA * NA + 255) / 256, 256, 0, stream>>>(draw, out_dist);

  // coord head
  launch_gemm<1, true>(stream, h_b, HD, 0, cpw1_b, HD, 0, cp_b1, nullptr,
                       nullptr, 0, hc, HD, nullptr, 0, 0, NA, HD, HD, 1);
  coord_kernel<<<(NA * 3 + 255) / 256, 256, 0, stream>>>(hc, cp_w2, cp_b2, out_x);
}
